// AdiabaticTDDFTNN_33354716021421
// MI455X (gfx1250) — matrix operand order repaired and checked by reading, not compiled, not run on hardware
//
#include <hip/hip_runtime.h>
#include <hip/hip_fp16.h>
#include <hip/hip_bf16.h>

typedef __attribute__((ext_vector_type(16))) _Float16 v16h;
typedef __attribute__((ext_vector_type(8)))  _Float16 v8h;
typedef __attribute__((ext_vector_type(8)))  float    v8f;

typedef union { v16h v; v8h h[2]; } pack16;

#define B_   128
#define L_   128
#define T_   128
#define HC_  40
#define HCP  48    // padded channels (3 x 16)
#define KTOT 256   // tap-major im2col K: kk*48 + cin, padded 240 -> 256 (8 k-steps of 32)
#define KROW 264   // Wim storage row stride in halves (528B = 132 dw == 4 mod 64 -> bank spread)
#define ACTW 56    // act_t storage row stride in halves (112B = 28 dw, gcd(28,64)=4 -> conflict-free)
#define PROW 132   // psi row stride in floats (528B: float4-aligned rows, conflict-free columns)

// ---- LDS layout (bytes) ----
#define OFF_PSI_R   0
#define OFF_PSI_I   (OFF_PSI_R + 128*PROW*4)        // 67584
#define OFF_WIM1    (OFF_PSI_I + 128*PROW*4)        // 135168 (16B aligned)
#define OFF_WIM2    (OFF_WIM1 + HCP*KROW*2)         // 160512
#define OFF_ACTA    (OFF_WIM2 + HCP*KROW*2)         // 185856
#define OFF_ACTB    (OFF_ACTA + L_*ACTW*2)          // 200192
#define OFF_W0      (OFF_ACTB + L_*ACTW*2)          // 214528
#define OFF_W3      (OFF_W0 + 200*4)
#define OFF_B0      (OFF_W3 + 200*4)
#define OFF_B1P     (OFF_B0 + 40*4)
#define OFF_B2P     (OFF_B1P + 48*4)
#define OFF_B3      (OFF_B2P + 48*4)
#define OFF_ZARR    (OFF_B3 + 4)
#define OFF_F1A     (OFF_ZARR + 128*4)
#define OFF_F4A     (OFF_F1A + 128*4)
#define OFF_ZP      (OFF_F4A + 128*4)               // 2*128 deterministic partials
#define OFF_MAGP    (OFF_ZP + 2*128*4)              // 8*128 per-wave partials
#define LDS_BYTES   (OFF_MAGP + 8*128*4)            // ~223 KB, fits 320 KB WGP LDS

// 40->40 circular conv as tap-major im2col GEMM on v_wmma_f32_16x16x32_f16.
// Wave w owns N columns 16w..16w+15; 3 M-tiles accumulate in registers.
// actIn/actOut are transposed activation planes: act[position l][channel c].
__device__ __forceinline__ void conv_wmma(const _Float16* __restrict__ Wim,
                                          const _Float16* __restrict__ actIn,
                                          _Float16* __restrict__ actOut,
                                          const float* __restrict__ biasP,
                                          int wave, int lane)
{
    const int nl   = wave * 16 + (lane & 15);   // this lane's N column (B and C/D layouts)
    const int hiK  = lane >> 4;                 // lane half selects K sub-range
    const int mrow = lane & 15;
    v8f acc0 = {}, acc1 = {}, acc2 = {};
    #pragma unroll
    for (int kt = 0; kt < 8; ++kt) {
        const int kb = kt * 32;
        // B fragment (32x16 K x N): half j -> K = kb + hiK*16 + j : 16 contiguous K.
        // Tap-major K = kk*48 + cin; a 16-aligned run never crosses a tap block (48 = 3*16),
        // so it is 16 contiguous halves of one act_t row -> two ds_load_b128.
        const int Kr = kb + hiK * 16;
        const int kk = Kr / 48;                 // tap index (K >= 240 -> kk 5, weights are zero)
        const int co = Kr - kk * 48;            // channel offset within tap block (0/16/32)
        const v8h* bp = (const v8h*)(actIn + ((nl + kk + 126) & 127) * ACTW + co);
        pack16 bfr;
        bfr.h[0] = bp[0];
        bfr.h[1] = bp[1];
        // A fragments (16x32 M x K): per lane two contiguous 8-half runs:
        // halves 0-15 -> K = kb + hiK*16 + (0..15)
        pack16 a0, a1, a2;
        {
            const _Float16* w0 = Wim + (mrow)      * KROW + kb + hiK * 16;
            const _Float16* w1 = Wim + (16 + mrow) * KROW + kb + hiK * 16;
            const _Float16* w2 = Wim + (32 + mrow) * KROW + kb + hiK * 16;
            a0.h[0] = *(const v8h*)(w0);  a0.h[1] = *(const v8h*)(w0 + 8);
            a1.h[0] = *(const v8h*)(w1);  a1.h[1] = *(const v8h*)(w1 + 8);
            a2.h[0] = *(const v8h*)(w2);  a2.h[1] = *(const v8h*)(w2 + 8);
        }
        acc0 = __builtin_amdgcn_wmma_f32_16x16x32_f16(false, a0.v, false, bfr.v, (short)0, acc0, false, false);
        acc1 = __builtin_amdgcn_wmma_f32_16x16x32_f16(false, a1.v, false, bfr.v, (short)0, acc1, false, false);
        acc2 = __builtin_amdgcn_wmma_f32_16x16x32_f16(false, a2.v, false, bfr.v, (short)0, acc2, false, false);
    }
    // D layout: element r -> M = mt*16 + hiK*8 + r (8 consecutive channels per lane).
    // Bias + ReLU, pack to v8h, one ds_store_b128 per M-tile into transposed act plane.
    // Padded channels (M>=40) get relu(0+0)=0, keeping next layer's padded-K reads zero.
    #pragma unroll
    for (int mt = 0; mt < 3; ++mt) {
        const v8f& acc = (mt == 0) ? acc0 : (mt == 1) ? acc1 : acc2;
        const int Mb = mt * 16 + hiK * 8;
        v8h o;
        #pragma unroll
        for (int r = 0; r < 8; ++r)
            o[r] = (_Float16)fmaxf(acc[r] + biasP[Mb + r], 0.f);
        *(v8h*)(actOut + nl * ACTW + Mb) = o;
    }
}

__global__ __launch_bounds__(256)
void adia_kernel(const float* __restrict__ h,
                 const float* __restrict__ Wc0, const float* __restrict__ bc0,
                 const float* __restrict__ Wc1, const float* __restrict__ bc1,
                 const float* __restrict__ Wc2, const float* __restrict__ bc2,
                 const float* __restrict__ Wc3, const float* __restrict__ bc3,
                 float* __restrict__ out)
{
    extern __shared__ char smem[];
    float*    psi_r = (float*)(smem + OFF_PSI_R);
    float*    psi_i = (float*)(smem + OFF_PSI_I);
    _Float16* Wim1  = (_Float16*)(smem + OFF_WIM1);
    _Float16* Wim2  = (_Float16*)(smem + OFF_WIM2);
    _Float16* actA  = (_Float16*)(smem + OFF_ACTA);
    _Float16* actB  = (_Float16*)(smem + OFF_ACTB);
    float*    W0    = (float*)(smem + OFF_W0);
    float*    W3    = (float*)(smem + OFF_W3);
    float*    b0    = (float*)(smem + OFF_B0);
    float*    b1p   = (float*)(smem + OFF_B1P);
    float*    b2p   = (float*)(smem + OFF_B2P);
    float*    b3s   = (float*)(smem + OFF_B3);
    float*    zarr  = (float*)(smem + OFF_ZARR);
    float*    f1a   = (float*)(smem + OFF_F1A);
    float*    f4a   = (float*)(smem + OFF_F4A);
    float*    zp    = (float*)(smem + OFF_ZP);
    float*    magp  = (float*)(smem + OFF_MAGP);

    const int tid  = threadIdx.x;
    const int a    = blockIdx.x;
    const int wave = tid >> 5;
    const int lane = tid & 31;
    const float DT = (float)(6.4 / 127.0);   // linspace(0,6.4,128) spacing

    // ---- setup: psi0, tap-major im2col weights (f16), biases, z0 output row ----
    for (int idx = tid; idx < 128 * 128; idx += 256) {
        int m = idx >> 7, l = idx & 127;
        psi_r[m * PROW + l] = (m == 0) ? 0.70710678118654752f : 0.0f;
        psi_i[m * PROW + l] = 0.0f;
    }
    for (int idx = tid; idx < HCP * KROW; idx += 256) {
        int M = idx / KROW, K = idx - M * KROW;
        int kk = K / 48, cin = K - kk * 48;
        float w1 = 0.f, w2 = 0.f;
        if (M < HC_ && kk < 5 && cin < HC_) {
            w1 = Wc1[(M * HC_ + cin) * 5 + kk];
            w2 = Wc2[(M * HC_ + cin) * 5 + kk];
        }
        Wim1[idx] = (_Float16)w1;
        Wim2[idx] = (_Float16)w2;
    }
    for (int idx = tid; idx < 200; idx += 256) { W0[idx] = Wc0[idx]; W3[idx] = Wc3[idx]; }
    if (tid < 48) { b1p[tid] = (tid < HC_) ? bc1[tid] : 0.f; b2p[tid] = (tid < HC_) ? bc2[tid] : 0.f; }
    if (tid < 40) b0[tid] = bc0[tid];
    if (tid == 0) b3s[0] = bc3[0];
    // z0 = 1 - 2*sum|psi0|^2 = 0 everywhere -> z_ml[a][T-1][l] = 0
    if (tid < 128) out[((size_t)a * T_ + (T_ - 1)) * L_ + tid] = 0.f;
    __syncthreads();

    // ================= time scan: 127 sequential steps =================
    for (int t = 0; t < T_ - 1; ++t) {
        // --- z[l] = 1 - 2 Re sum_m conj(psi[m,l]) psi[l,m] ---
        // Row direction via float4 (aligned, PROW=132); column direction coalesced across lanes.
        {
            int l = tid & 127, half = tid >> 7;
            const float* rr = psi_r + l * PROW;
            const float* ri = psi_i + l * PROW;
            float s = 0.f;
            for (int m0 = half * 64; m0 < half * 64 + 64; m0 += 4) {
                float4 xr = *(const float4*)(rr + m0);
                float4 xi = *(const float4*)(ri + m0);
                s += xr.x * psi_r[(m0 + 0) * PROW + l] + xi.x * psi_i[(m0 + 0) * PROW + l];
                s += xr.y * psi_r[(m0 + 1) * PROW + l] + xi.y * psi_i[(m0 + 1) * PROW + l];
                s += xr.z * psi_r[(m0 + 2) * PROW + l] + xi.z * psi_i[(m0 + 2) * PROW + l];
                s += xr.w * psi_r[(m0 + 3) * PROW + l] + xi.w * psi_i[(m0 + 3) * PROW + l];
            }
            zp[half * 128 + l] = s;   // deterministic 2-way partial
        }
        __syncthreads();
        if (tid < 128) zarr[tid] = 1.f - 2.f * (zp[tid] + zp[128 + tid]);
        __syncthreads();

        // --- layer 0: 1 -> 40, circular, ReLU, store f16 transposed (pad channels = 0) ---
        for (int idx = tid; idx < HCP * L_; idx += 256) {
            int c = idx >> 7, l = idx & 127;
            float y = 0.f;
            if (c < HC_) {
                y = b0[c];
                #pragma unroll
                for (int k = 0; k < 5; ++k)
                    y += W0[c * 5 + k] * zarr[(l + k + 126) & 127];
                y = fmaxf(y, 0.f);
            }
            actA[l * ACTW + c] = (_Float16)y;
        }
        __syncthreads();

        // --- layers 1 & 2: 40 -> 40 WMMA GEMMs (uniform across waves, EXEC all-1s) ---
        conv_wmma(Wim1, actA, actB, b1p, wave, lane);
        __syncthreads();
        conv_wmma(Wim2, actB, actA, b2p, wave, lane);
        __syncthreads();

        // --- layer 3: 40 -> 1 (no ReLU), then field arrays f1 = v+h[t], f4 = v+h[t+1] ---
        if (tid < 128) {
            int l = tid;
            float y = b3s[0];
            for (int cin = 0; cin < HC_; ++cin) {
                #pragma unroll
                for (int k = 0; k < 5; ++k)
                    y += W3[cin * 5 + k] * (float)actA[((l + k + 126) & 127) * ACTW + cin];
            }
            f1a[l] = y + h[((size_t)a * T_ + t)     * L_ + l];
            f4a[l] = y + h[((size_t)a * T_ + t + 1) * L_ + l];
        }
        __syncthreads();

        // --- RK4 in registers: wave owns rows wave*16..+15, lane owns cols lane*4..+3 ---
        {
            const int i0 = lane * 4;
            const int prevL = (lane + 31) & 31, nextL = (lane + 1) & 31;
            float fl1[4], fl4[4], magacc[4] = {0, 0, 0, 0};
            #pragma unroll
            for (int c = 0; c < 4; ++c) { fl1[c] = f1a[i0 + c]; fl4[c] = f4a[i0 + c]; }
            for (int r = 0; r < 16; ++r) {
                int m = wave * 16 + r;
                float4 p4r = *(const float4*)(psi_r + m * PROW + i0);   // aligned b128
                float4 p4i = *(const float4*)(psi_i + m * PROW + i0);
                float pr[4] = {p4r.x, p4r.y, p4r.z, p4r.w};
                float pi[4] = {p4i.x, p4i.y, p4i.z, p4i.w};
                float sr[4], si[4], ar[4] = {0,0,0,0}, ai[4] = {0,0,0,0};
                #pragma unroll
                for (int c = 0; c < 4; ++c) { sr[c] = pr[c]; si[c] = pi[c]; }
                // stages: k = -i*(H s), H = circulant lap + diag(f); k_re = H s_im, k_im = -H s_re
                #pragma unroll
                for (int s = 0; s < 4; ++s) {
                    const float wgt = (s == 0) ? DT : ((s == 3) ? 1.f : 2.f);   // dt,2,2,1
                    const float cs  = (s < 2) ? DT * 0.5f : DT;                 // dt/2,dt/2,dt
                    const float* f  = (s == 3) ? fl4 : fl1;                     // f2 == f1
                    float pm_r = __shfl(sr[3], prevL), pm_i = __shfl(si[3], prevL);
                    float pp_r = __shfl(sr[0], nextL), pp_i = __shfl(si[0], nextL);
                    float kr[4], ki[4];
                    #pragma unroll
                    for (int c = 0; c < 4; ++c) {
                        float xm_r = (c == 0) ? pm_r : sr[c - 1];
                        float xm_i = (c == 0) ? pm_i : si[c - 1];
                        float xp_r = (c == 3) ? pp_r : sr[c + 1];
                        float xp_i = (c == 3) ? pp_i : si[c + 1];
                        float Hr = 2.f * sr[c] - xm_r - xp_r + f[c] * sr[c];
                        float Hi = 2.f * si[c] - xm_i - xp_i + f[c] * si[c];
                        kr[c] = Hi; ki[c] = -Hr;
                    }
                    #pragma unroll
                    for (int c = 0; c < 4; ++c) { ar[c] += wgt * kr[c]; ai[c] += wgt * ki[c]; }
                    if (s < 3) {
                        #pragma unroll
                        for (int c = 0; c < 4; ++c) { sr[c] = pr[c] + cs * kr[c]; si[c] = pi[c] + cs * ki[c]; }
                    }
                }
                float4 n4r, n4i;
                float nr[4], ni[4];
                #pragma unroll
                for (int c = 0; c < 4; ++c) {
                    nr[c] = pr[c] + (DT / 6.f) * ar[c];
                    ni[c] = pi[c] + (DT / 6.f) * ai[c];
                    magacc[c] += nr[c] * nr[c] + ni[c] * ni[c];
                }
                n4r.x = nr[0]; n4r.y = nr[1]; n4r.z = nr[2]; n4r.w = nr[3];
                n4i.x = ni[0]; n4i.y = ni[1]; n4i.z = ni[2]; n4i.w = ni[3];
                *(float4*)(psi_r + m * PROW + i0) = n4r;                // aligned b128
                *(float4*)(psi_i + m * PROW + i0) = n4i;
            }
            #pragma unroll
            for (int c = 0; c < 4; ++c) magp[wave * 128 + i0 + c] = magacc[c];   // per-wave partial
        }
        __syncthreads();

        // --- mag output: fixed-order 8-way reduction; z_ml[a][126-t][l] ---
        if (tid < 128) {
            float s = 0.f;
            #pragma unroll
            for (int w = 0; w < 8; ++w) s += magp[w * 128 + tid];
            out[((size_t)a * T_ + (126 - t)) * L_ + tid] = 1.f - 2.f * s;
        }
        __syncthreads();
    }

    // ---- final psi (complex64 interleaved) after the z_ml block: float2 stores ----
    const size_t pbase = (size_t)B_ * T_ * L_ + (size_t)2 * a * L_ * L_;
    for (int idx = tid; idx < L_ * L_; idx += 256) {
        int m = idx >> 7, l = idx & 127;
        float2 c2;
        c2.x = psi_r[m * PROW + l];
        c2.y = psi_i[m * PROW + l];
        *(float2*)(out + pbase + 2 * idx) = c2;
    }
}

extern "C" void kernel_launch(void* const* d_in, const int* in_sizes, int n_in,
                              void* d_out, int out_size, void* d_ws, size_t ws_size,
                              hipStream_t stream) {
    const float* h   = (const float*)d_in[0];
    const float* Wc0 = (const float*)d_in[1];
    const float* bc0 = (const float*)d_in[2];
    const float* Wc1 = (const float*)d_in[3];
    const float* bc1 = (const float*)d_in[4];
    const float* Wc2 = (const float*)d_in[5];
    const float* bc2 = (const float*)d_in[6];
    const float* Wc3 = (const float*)d_in[7];
    const float* bc3 = (const float*)d_in[8];
    adia_kernel<<<B_, 256, LDS_BYTES, stream>>>(h, Wc0, bc0, Wc1, bc1, Wc2, bc2, Wc3, bc3,
                                                (float*)d_out);
    (void)in_sizes; (void)n_in; (void)out_size; (void)d_ws; (void)ws_size;
}
